// DeTGC_59399397703719
// MI455X (gfx1250) — matrix operand order repaired and checked by reading, not compiled, not run on hardware
//
#include <hip/hip_runtime.h>
#include <hip/hip_bf16.h>

typedef __attribute__((ext_vector_type(16))) __bf16 v16bf;
typedef __attribute__((ext_vector_type(8)))  __bf16 v8bf;
typedef __attribute__((ext_vector_type(8)))  float  v8f;
typedef __attribute__((ext_vector_type(4)))  float  v4f;

#define N_   32
#define C_   64
#define T_   300
#define V_   25
#define ETA_ 9
#define OUT_ 64
#define K_TOT   (C_ * ETA_)      // 576
#define K_PHASE 288              // half of K staged per phase
#define KC      32               // K per WMMA
#define PITCH   296              // K_PHASE + 8 pad (multiple of 8 bf16 -> 16B aligned rows)
#define COLS_PER_BLOCK 32
#define NCOL_TOT (T_ * V_)       // 7500

// round-to-nearest-even f32 -> bf16
__device__ __forceinline__ __bf16 f2bf(float f) {
  unsigned u = __float_as_uint(f);
  u += 0x7fffu + ((u >> 16) & 1u);
  unsigned short h = (unsigned short)(u >> 16);
  __bf16 r;
  __builtin_memcpy(&r, &h, sizeof(r));
  return r;
}

__global__ __launch_bounds__(256)
void detgc_wmma_kernel(const float* __restrict__ x, const float* __restrict__ tr,
                       const float* __restrict__ W, const float* __restrict__ bias,
                       const float* __restrict__ gamma, const float* __restrict__ beta,
                       const float* __restrict__ rmean, const float* __restrict__ rvar,
                       float* __restrict__ out)
{
  __shared__ __bf16 Alds[OUT_ * PITCH];            // [o 0..63][k 0..287] row-major
  __shared__ __bf16 Blds[COLS_PER_BLOCK * PITCH];  // [col 0..31][k 0..287] K-major per col
  __shared__ float  sScale[OUT_];
  __shared__ float  sShift[OUT_];
  __shared__ float  sAlpha[ETA_];
  __shared__ int    sDoff[ETA_];

  const int tid  = threadIdx.x;
  const int lane = tid & 31;
  const int wave = tid >> 5;     // 0..7
  const int wo   = wave & 3;     // o-tile  (16 rows each, covers OUT=64)
  const int wc   = wave >> 2;    // col-tile (16 cols each, covers 32)
  const int n    = blockIdx.y;
  const int colBase = blockIdx.x * COLS_PER_BLOCK;

  // Folded BN epilogue params: y = gemm * scale[o] + shift[o]
  if (tid < OUT_) {
    float inv = rsqrtf(rvar[tid] + 1e-5f);
    float sc  = gamma[tid] * inv;
    sScale[tid] = sc;
    sShift[tid] = beta[tid] + (bias[tid] - rmean[tid]) * sc;
  }
  // Per-eta interp constants: alpha and integer displacement
  if (tid < ETA_) {
    float trv = tr[tid];
    float d   = floorf(trv);
    sDoff[tid]  = (int)d;
    sAlpha[tid] = d + 1.0f - trv;    // alpha = (tdn+1) - t
  }
  __syncthreads();

  const float* xN = x + (size_t)n * C_ * T_ * V_;

  // --- B staging assignment: fixed column per thread, wave-uniform k ---
  const int bcol  = tid & 31;                              // local col
  const int bkrow = tid >> 5;                              // 0..7; k = bkrow + 8*i
  const int bcolG = min(colBase + bcol, NCOL_TOT - 1);     // clamp ragged tail
  const int lt    = bcolG / V_;
  const int lv    = bcolG - lt * V_;

  // --- A staging assignment: thread -> (o row, 72 contiguous k as float4) ---
  const int arow   = tid >> 2;                             // 0..63
  const int akbase = (tid & 3) * (K_PHASE / 4);            // 0,72,144,216
  const float* wRow = W + (size_t)arow * K_TOT;

  // fragment addressing (ISA VGPR layouts)
  const int fm    = lane & 15;
  const int khalf = lane >> 4;
  const __bf16* aRowP = &Alds[(wo * 16 + fm) * PITCH];
  const __bf16* bColP = &Blds[(wc * 16 + fm) * PITCH];

  v8f acc = {};

  for (int p = 0; p < 2; ++p) {
    const int kPh = p * K_PHASE;
    __syncthreads();   // protect LDS from previous phase's fragment reads

    // ---- Stage W half: 64 x 288 f32 -> bf16, float4-vectorized ----
    #pragma unroll 4
    for (int j = 0; j < K_PHASE / 4; j += 4) {
      v4f wv = *(const v4f*)&wRow[kPh + akbase + j];
      __bf16* dst = &Alds[arow * PITCH + akbase + j];
      dst[0] = f2bf(wv.x); dst[1] = f2bf(wv.y);
      dst[2] = f2bf(wv.z); dst[3] = f2bf(wv.w);
    }

    // ---- Stage xi half: 32 cols x 288 k, wave-uniform (c,e) per step ----
    #pragma unroll 4
    for (int i = 0; i < K_PHASE / 8; ++i) {                // 36 steps
      int kloc = bkrow + (i << 3);
      int k    = kPh + kloc;
      int c    = k / ETA_;
      int e    = k - c * ETA_;
      int de   = sDoff[e];
      float a  = sAlpha[e];
      int i1 = min(max(lt + de,     0), T_ - 1);
      int i2 = min(max(lt + de + 1, 0), T_ - 1);
      const float* xc = xN + (size_t)c * T_ * V_;
      float val = a * xc[i1 * V_ + lv] + (1.0f - a) * xc[i2 * V_ + lv];
      Blds[bcol * PITCH + kloc] = f2bf(val);
    }
    __syncthreads();

    // ---- 9 back-to-back WMMAs straight out of LDS ----
    #pragma unroll
    for (int j = 0; j < K_PHASE / KC; ++j) {
      const int kb = j * KC;
      // A: elems 0-7 <- K=kb+khalf*8+0..7 ; elems 8-15 <- K=kb+16+khalf*8+0..7
      v8bf alo = *(const v8bf*)&aRowP[kb + khalf * 8];
      v8bf ahi = *(const v8bf*)&aRowP[kb + 16 + khalf * 8];
      v16bf afrag = __builtin_shufflevector(alo, ahi,
          0,1,2,3,4,5,6,7,8,9,10,11,12,13,14,15);
      // B: col = lane&15, elems 0-7 <- K=kb+khalf*8+0..7 ; elems 8-15 <- K=kb+16+khalf*8+0..7
      v8bf blo = *(const v8bf*)&bColP[kb + khalf * 8];
      v8bf bhi = *(const v8bf*)&bColP[kb + 16 + khalf * 8];
      v16bf bfrag = __builtin_shufflevector(blo, bhi,
          0,1,2,3,4,5,6,7,8,9,10,11,12,13,14,15);

      acc = __builtin_amdgcn_wmma_f32_16x16x32_bf16(
          /*neg_a=*/false, afrag, /*neg_b=*/false, bfrag,
          /*c_mod=*/(short)0, acc, /*reuse_a=*/false, /*reuse_b=*/false);
    }
  }

  // ---- Epilogue: BN fold + store. C/D: VGPR r -> M=(lane>>4)*8+r, N=lane&15 ----
  const int m0   = (lane >> 4) * 8;
  const int ncol = lane & 15;
  const int colG = colBase + wc * 16 + ncol;
  if (colG < NCOL_TOT) {
    const int t = colG / V_;
    const int v = colG - t * V_;
    float* outN = out + (size_t)n * OUT_ * (T_ * V_) + t * V_ + v;
    #pragma unroll
    for (int r = 0; r < 8; ++r) {
      int o = wo * 16 + m0 + r;
      outN[(size_t)o * (T_ * V_)] = acc[r] * sScale[o] + sShift[o];
    }
  }
}

extern "C" void kernel_launch(void* const* d_in, const int* in_sizes, int n_in,
                              void* d_out, int out_size, void* d_ws, size_t ws_size,
                              hipStream_t stream) {
  const float* x     = (const float*)d_in[0];
  const float* tr    = (const float*)d_in[1];
  const float* W     = (const float*)d_in[2];
  const float* b     = (const float*)d_in[3];
  const float* gamma = (const float*)d_in[4];
  const float* beta  = (const float*)d_in[5];
  const float* rmean = (const float*)d_in[6];
  const float* rvar  = (const float*)d_in[7];
  float* out = (float*)d_out;

  dim3 grid((NCOL_TOT + COLS_PER_BLOCK - 1) / COLS_PER_BLOCK, N_);  // 235 x 32
  detgc_wmma_kernel<<<grid, 256, 0, stream>>>(x, tr, W, b, gamma, beta,
                                              rmean, rvar, out);
}
